// AE_RNN_4432406249580
// MI455X (gfx1250) — compile-verified, matrix operand order checked by reading, not run on hardware
//
#include <hip/hip_runtime.h>

#define DEV __device__ __forceinline__

typedef _Float16 f16;
typedef __attribute__((ext_vector_type(16))) _Float16 v16h;
typedef __attribute__((ext_vector_type(8)))  _Float16 v8h;
typedef __attribute__((ext_vector_type(8)))  float    v8f;

// Problem dims
constexpr int B_  = 1024;
constexpr int T_  = 1024;
constexpr int UD_ = 16;
constexpr int YD_ = 16;
constexpr int ZD_ = 16;
constexpr int HD_ = 128;

// Launch shape
constexpr int BM   = 16;        // batch rows per workgroup (WMMA M)
constexpr int NW   = 8;         // waves per workgroup
constexpr int NTHR = NW * 32;   // 256 threads

// ---------------- LDS layout (units: halfs for f16 region) ----------------
constexpr int O_W_PHIU0 = 0;                         // 128 x 32  (K=16 padded)
constexpr int O_W_PHIU1 = O_W_PHIU0 + 128 * 32;      // 128 x 128
constexpr int O_W_DYNN0 = O_W_PHIU1 + 128 * 128;     // 128 x 256
constexpr int O_W_DYNN1 = O_W_DYNN0 + 128 * 256;     // 128 x 128
constexpr int O_W_XM    = O_W_DYNN1 + 128 * 128;     // 16 x 128
constexpr int O_W_XLV   = O_W_XM    + 16 * 128;      // 16 x 128
constexpr int O_W_PHIX0 = O_W_XLV   + 16 * 128;      // 128 x 32
constexpr int O_W_PHIX1 = O_W_PHIX0 + 128 * 32;      // 128 x 128
constexpr int O_W_MENN0 = O_W_PHIX1 + 128 * 128;     // 128 x 128
constexpr int O_W_MENN1 = O_W_MENN0 + 128 * 128;     // 16 x 128
constexpr int O_W_C     = O_W_MENN1 + 16 * 128;      // 16 x 32 (K=16 zero-padded)
// GRU n-gate recurrent weights (both layers) live in LDS to cap VGPR usage
constexpr int O_W_GRUN  = O_W_C     + 16 * 32;       // 2 x 128 x 128
// activations
constexpr int O_UA      = O_W_GRUN  + 2 * 128 * 128; // 16 x 32 u_t tile (K padded w/ zeros)
constexpr int O_DYNIN   = O_UA      + 16 * 32;       // 16 x 256 = [phi_u_t | h1]
constexpr int O_H0      = O_DYNIN   + 16 * 256;      // 16 x 128 GRU layer-0 hidden
constexpr int O_HBUFA   = O_H0      + 16 * 128;      // 16 x 128 scratch hidden (relu outputs)
constexpr int O_DPHI    = O_HBUFA   + 16 * 128;      // 16 x 128 dynn output
constexpr int O_ZBUF    = O_DPHI    + 16 * 128;      // 16 x 32  [x_mean | x_logvar]
constexpr int O_PHIX    = O_ZBUF    + 16 * 32;       // 16 x 128 phi_x output
constexpr int O_MH      = O_PHIX    + 16 * 128;      // 16 x 128 menn hidden
constexpr int H_TOTAL   = O_MH      + 16 * 128;      // total halfs = 161280

// float bias region (units: floats), placed after the f16 region
constexpr int BF_PHIU0 = 0;
constexpr int BF_PHIU1 = BF_PHIU0 + 128;
constexpr int BF_DYNN0 = BF_PHIU1 + 128;
constexpr int BF_DYNN1 = BF_DYNN0 + 128;
constexpr int BF_XM    = BF_DYNN1 + 128;
constexpr int BF_XLV   = BF_XM    + 16;
constexpr int BF_PHIX0 = BF_XLV   + 16;
constexpr int BF_PHIX1 = BF_PHIX0 + 128;
constexpr int BF_MENN0 = BF_PHIX1 + 128;
constexpr int BF_MENN1 = BF_MENN0 + 128;
constexpr int BF_TOTAL = BF_MENN1 + 16;              // 944 floats

constexpr int SMEM_BYTES = H_TOTAL * 2 + BF_TOTAL * 4;  // 326336 B <= 327680 B (320 KB)

// ---------------- WMMA helpers ----------------
DEV v8f wmma16(v16h a, v16h b, v8f c) {
    // D = A(16x32 f16) * B(32x16 f16) + C(16x16 f32)
    return __builtin_amdgcn_wmma_f32_16x16x32_f16(false, a, false, b, (short)0, c,
                                                  false, false);
}

// Load one 16x32 fragment (A: lane%16 = M row; B: lane%16 = N row of row-major (out,in) W)
// Element e<8  -> col (hi + e), element e>=8 -> col (16 + hi + e-8), hi = (lane>=16)*8.
DEV v16h frag_lds(const f16* base, int stride, int lane) {
    const int r  = lane & 15;
    const int hi = (lane & 16) ? 8 : 0;
    const f16* p = base + r * stride + hi;
    union { v16h v; v8h h[2]; } u;
    u.h[0] = *(const v8h*)(p);
    u.h[1] = *(const v8h*)(p + 16);
    return u.v;
}

// Same fragment shape, loaded from row-major f32 in global (weights at init)
DEV v16h frag_gf32(const float* base, int stride, int lane) {
    const int r  = lane & 15;
    const int hi = (lane & 16) ? 8 : 0;
    const float* p = base + (size_t)r * stride + hi;
    v16h o;
#pragma unroll
    for (int e = 0; e < 8; ++e) o[e] = (f16)p[e];
#pragma unroll
    for (int e = 0; e < 8; ++e) o[8 + e] = (f16)p[16 + e];
    return o;
}

// One 16x16 output tile: acc = bias; acc += A(16,32k) * W^T tile; optional ReLU; store f16
template <int KT, bool RELU>
DEV void dense_tile(const f16* A, int astr, const f16* Wrow, int wstr,
                    const float* bias_j, f16* out, int ostr, int ncol0, int lane) {
    const int n  = lane & 15;
    const int mb = (lane & 16) ? 8 : 0;
    const float bv = bias_j[n];
    v8f acc;
#pragma unroll
    for (int r = 0; r < 8; ++r) acc[r] = bv;
#pragma unroll
    for (int kt = 0; kt < KT; ++kt)
        acc = wmma16(frag_lds(A + kt * 32, astr, lane),
                     frag_lds(Wrow + kt * 32, wstr, lane), acc);
#pragma unroll
    for (int r = 0; r < 8; ++r) {
        float v = acc[r];
        if (RELU) v = fmaxf(v, 0.f);
        out[(mb + r) * ostr + ncol0 + n] = (f16)v;
    }
}

// One GRU layer step for this wave's 16 hidden columns.
// r/z gates: Wih/Whh register-resident; n gate: Wih register-resident, Whh from LDS.
DEV void gru_layer(const f16* xbuf, int xstr, f16* hbuf, int hstr,
                   const v16h (&Bih)[3][4], const v16h (&Bhh)[2][4],
                   const f16* WhhN, int lane, int w) {
    v8f ar, az, ani, anh;
#pragma unroll
    for (int r = 0; r < 8; ++r) { ar[r] = 0.f; az[r] = 0.f; ani[r] = 0.f; anh[r] = 0.f; }
#pragma unroll
    for (int kt = 0; kt < 4; ++kt) {
        v16h ax = frag_lds(xbuf + kt * 32, xstr, lane);
        v16h ah = frag_lds(hbuf + kt * 32, hstr, lane);
        ar  = wmma16(ax, Bih[0][kt], ar);
        ar  = wmma16(ah, Bhh[0][kt], ar);
        az  = wmma16(ax, Bih[1][kt], az);
        az  = wmma16(ah, Bhh[1][kt], az);
        ani = wmma16(ax, Bih[2][kt], ani);
        anh = wmma16(ah, frag_lds(WhhN + kt * 32, HD_, lane), anh);
    }
    const int n = lane & 15, mb = (lane & 16) ? 8 : 0, col = w * 16 + n;
    float hnew[8];
#pragma unroll
    for (int r = 0; r < 8; ++r) {
        float hold = (float)hbuf[(mb + r) * hstr + col];
        float rg = 1.f / (1.f + __expf(-ar[r]));
        float zg = 1.f / (1.f + __expf(-az[r]));
        float ng = tanhf(ani[r] + rg * anh[r]);
        hnew[r] = (1.f - zg) * ng + zg * hold;
    }
    __syncthreads();   // all reads of h done before overwriting
#pragma unroll
    for (int r = 0; r < 8; ++r)
        hbuf[(mb + r) * hstr + col] = (f16)hnew[r];
    __syncthreads();
}

// ---------------- staging helpers ----------------
DEV void stage_w16(f16* dst, const float* src, int N, int K, int Kpad, int tid) {
    for (int i = tid; i < N * Kpad; i += NTHR) {
        int r = i / Kpad, c = i - r * Kpad;
        dst[i] = (c < K) ? (f16)src[r * K + c] : (f16)0.f;
    }
}
DEV void stage_bf(float* dst, const float* src, int n, int tid) {
    for (int i = tid; i < n; i += NTHR) dst[i] = src[i];
}

// ---------------- main persistent kernel ----------------
__global__ void __launch_bounds__(NTHR, 1)
vrnn_kernel(const float* __restrict__ u, const float* __restrict__ y,
            const float* __restrict__ h0,
            const float* pw0, const float* pb0, const float* pw1, const float* pb1,
            const float* dw0, const float* db0, const float* dw1, const float* db1,
            const float* xmw, const float* xmb, const float* xlw, const float* xlb,
            const float* qw0, const float* qb0, const float* qw1, const float* qb1,
            const float* mw0, const float* mb0, const float* mw1, const float* mb1,
            const float* wih, const float* whh, const float* Cm,
            float* __restrict__ out) {
    extern __shared__ char smem_raw[];
    f16*   sm  = (f16*)smem_raw;
    float* smf = (float*)(smem_raw + (size_t)H_TOTAL * 2);

    const int tid   = threadIdx.x;
    const int lane  = tid & 31;
    const int w     = tid >> 5;           // wave id: owns hidden columns [16w,16w+16)
    const int bbase = blockIdx.x * BM;

    // ---- one-time staging: weights -> LDS f16 (K zero-padded), biases -> LDS f32 ----
    stage_w16(sm + O_W_PHIU0, pw0, HD_, UD_,     32,      tid);
    stage_w16(sm + O_W_PHIU1, pw1, HD_, HD_,     HD_,     tid);
    stage_w16(sm + O_W_DYNN0, dw0, HD_, 2 * HD_, 2 * HD_, tid);
    stage_w16(sm + O_W_DYNN1, dw1, HD_, HD_,     HD_,     tid);
    stage_w16(sm + O_W_XM,    xmw, ZD_, HD_,     HD_,     tid);
    stage_w16(sm + O_W_XLV,   xlw, ZD_, HD_,     HD_,     tid);
    stage_w16(sm + O_W_PHIX0, qw0, HD_, 2 * ZD_, 2 * ZD_, tid);
    stage_w16(sm + O_W_PHIX1, qw1, HD_, HD_,     HD_,     tid);
    stage_w16(sm + O_W_MENN0, mw0, HD_, HD_,     HD_,     tid);
    stage_w16(sm + O_W_MENN1, mw1, YD_, HD_,     HD_,     tid);
    stage_w16(sm + O_W_C,     Cm,  YD_, ZD_,     32,      tid);   // zero-padded K

    // GRU n-gate Whh (rows j = l*384 + 256 + r) -> LDS f16, both layers
    for (int i = tid; i < 2 * HD_ * HD_; i += NTHR) {
        int l = i >> 14;              // / (128*128)
        int r = (i >> 7) & 127;
        int k = i & 127;
        sm[O_W_GRUN + i] = (f16)whh[((size_t)(l * 3 * HD_ + 2 * HD_ + r)) * HD_ + k];
    }

    stage_bf(smf + BF_PHIU0, pb0, HD_, tid);
    stage_bf(smf + BF_PHIU1, pb1, HD_, tid);
    stage_bf(smf + BF_DYNN0, db0, HD_, tid);
    stage_bf(smf + BF_DYNN1, db1, HD_, tid);
    stage_bf(smf + BF_XM,    xmb, ZD_, tid);
    stage_bf(smf + BF_XLV,   xlb, ZD_, tid);
    stage_bf(smf + BF_PHIX0, qb0, HD_, tid);
    stage_bf(smf + BF_PHIX1, qb1, HD_, tid);
    stage_bf(smf + BF_MENN0, mb0, HD_, tid);
    stage_bf(smf + BF_MENN1, mb1, YD_, tid);

    // hidden state init: layer0 -> O_H0, layer1 -> DYNIN[:, 128:256]
    for (int i = tid; i < BM * HD_; i += NTHR) {
        int m = i >> 7, k = i & 127;
        sm[O_H0    + m * HD_  + k]       = (f16)h0[(size_t)(0 * B_ + bbase + m) * HD_ + k];
        sm[O_DYNIN + m * 256  + HD_ + k] = (f16)h0[(size_t)(1 * B_ + bbase + m) * HD_ + k];
    }
    // zero K-padding of u tile (cols 16..31 never written again)
    for (int i = tid; i < 256; i += NTHR) {
        int m = i >> 4, k = i & 15;
        sm[O_UA + m * 32 + 16 + k] = (f16)0.f;
    }
    __syncthreads();

    // ---- GRU weights: register-resident B fragments ----
    // Bih[l][gate 0..2][kt]; Bhh[l][gate 0..1][kt] (n-gate Whh comes from LDS).
    v16h Bih[2][3][4], Bhh[2][2][4];
#pragma unroll
    for (int l = 0; l < 2; ++l) {
#pragma unroll
        for (int g = 0; g < 3; ++g)
#pragma unroll
            for (int kt = 0; kt < 4; ++kt) {
                const size_t ro = (size_t)(l * 3 * HD_ + g * HD_ + w * 16) * HD_ + kt * 32;
                Bih[l][g][kt] = frag_gf32(wih + ro, HD_, lane);
            }
#pragma unroll
        for (int g = 0; g < 2; ++g)
#pragma unroll
            for (int kt = 0; kt < 4; ++kt) {
                const size_t ro = (size_t)(l * 3 * HD_ + g * HD_ + w * 16) * HD_ + kt * 32;
                Bhh[l][g][kt] = frag_gf32(whh + ro, HD_, lane);
            }
    }

    float lacc = 0.f;

    for (int t = 0; t < T_; ++t) {
        // 1. stage u_t tile (16 x 16 f32 -> f16, column t of (B,UD,T))
        {
            int m = tid >> 4, k = tid & 15;
            sm[O_UA + m * 32 + k] = (f16)u[((size_t)(bbase + m) * UD_ + k) * T_ + t];
        }
        __syncthreads();

        // 2. phi_u layer0 (K=16 padded to 32) -> HBUFA, ReLU
        dense_tile<1, true>(sm + O_UA, 32, sm + O_W_PHIU0 + (w * 16) * 32, 32,
                            smf + BF_PHIU0 + w * 16, sm + O_HBUFA, HD_, w * 16, lane);
        __syncthreads();
        // 3. phi_u layer1 -> DYNIN[:, 0:128]  (phi_u_t; also GRU layer-0 input)
        dense_tile<4, false>(sm + O_HBUFA, HD_, sm + O_W_PHIU1 + (w * 16) * HD_, HD_,
                             smf + BF_PHIU1 + w * 16, sm + O_DYNIN, 256, w * 16, lane);
        __syncthreads();

        // 4. dynn layer0 over concat [phi_u_t | h1_{t-1}] (K=256) -> HBUFA, ReLU
        dense_tile<8, true>(sm + O_DYNIN, 256, sm + O_W_DYNN0 + (w * 16) * 256, 256,
                            smf + BF_DYNN0 + w * 16, sm + O_HBUFA, HD_, w * 16, lane);
        __syncthreads();
        // 5. dynn layer1 -> DPHI
        dense_tile<4, false>(sm + O_HBUFA, HD_, sm + O_W_DYNN1 + (w * 16) * HD_, HD_,
                             smf + BF_DYNN1 + w * 16, sm + O_DPHI, HD_, w * 16, lane);
        __syncthreads();

        // 6. x_mean (wave 0) / x_logvar (wave 1) -> ZBUF = [x_mean | x_logvar]
        if (w == 0)
            dense_tile<4, false>(sm + O_DPHI, HD_, sm + O_W_XM, HD_,
                                 smf + BF_XM, sm + O_ZBUF, 32, 0, lane);
        else if (w == 1)
            dense_tile<4, false>(sm + O_DPHI, HD_, sm + O_W_XLV, HD_,
                                 smf + BF_XLV, sm + O_ZBUF, 32, 16, lane);
        __syncthreads();

        // 7. phi_x layer0 (K=32) -> HBUFA, ReLU
        dense_tile<1, true>(sm + O_ZBUF, 32, sm + O_W_PHIX0 + (w * 16) * 32, 32,
                            smf + BF_PHIX0 + w * 16, sm + O_HBUFA, HD_, w * 16, lane);
        __syncthreads();
        // 8. phi_x layer1 -> PHIX
        dense_tile<4, false>(sm + O_HBUFA, HD_, sm + O_W_PHIX1 + (w * 16) * HD_, HD_,
                             smf + BF_PHIX1 + w * 16, sm + O_PHIX, HD_, w * 16, lane);
        __syncthreads();
        // 9. menn layer0 -> MH, ReLU
        dense_tile<4, true>(sm + O_PHIX, HD_, sm + O_W_MENN0 + (w * 16) * HD_, HD_,
                            smf + BF_MENN0 + w * 16, sm + O_MH, HD_, w * 16, lane);
        __syncthreads();

        // 10. wave 0: y_hat = MH @ menn_w1^T + b1 + x_mean @ C^T ; accumulate loss
        if (w == 0) {
            const int n = lane & 15, mb = (lane & 16) ? 8 : 0;
            const float bv = smf[BF_MENN1 + n];
            v8f acc;
#pragma unroll
            for (int r = 0; r < 8; ++r) acc[r] = bv;
#pragma unroll
            for (int kt = 0; kt < 4; ++kt)
                acc = wmma16(frag_lds(sm + O_MH + kt * 32, HD_, lane),
                             frag_lds(sm + O_W_MENN1 + kt * 32, HD_, lane), acc);
            // physics term: A = ZBUF (x_logvar half multiplied by zero-padded C cols)
            acc = wmma16(frag_lds(sm + O_ZBUF, 32, lane),
                         frag_lds(sm + O_W_C, 32, lane), acc);
#pragma unroll
            for (int r = 0; r < 8; ++r) {
                int m = mb + r;
                float yv = y[((size_t)(bbase + m) * YD_ + n) * T_ + t];
                float d  = acc[r] - yv;
                lacc += d * d;
            }
        }

        // 11. GRU layer0: x = DYNIN[:,0:128] (phi_u_t), h = H0  (updates H0)
        gru_layer(sm + O_DYNIN, 256, sm + O_H0, HD_, Bih[0], Bhh[0],
                  sm + O_W_GRUN + 0 * HD_ * HD_ + (w * 16) * HD_, lane, w);
        // 12. GRU layer1: x = H0 (new), h = DYNIN[:,128:256]    (updates h1 slice)
        gru_layer(sm + O_H0, HD_, sm + O_DYNIN + HD_, 256, Bih[1], Bhh[1],
                  sm + O_W_GRUN + 1 * HD_ * HD_ + (w * 16) * HD_, lane, w);
        // gru_layer ends with a barrier -> safe to restage u_t next iteration
    }

    if (w == 0) {
#pragma unroll
        for (int off = 16; off > 0; off >>= 1)
            lacc += __shfl_down(lacc, off, 32);
        if (lane == 0) atomicAdd(out, lacc);
    }
}

__global__ void zero_out_kernel(float* out) {
    if (threadIdx.x == 0 && blockIdx.x == 0) out[0] = 0.f;
}

extern "C" void kernel_launch(void* const* d_in, const int* in_sizes, int n_in,
                              void* d_out, int out_size, void* d_ws, size_t ws_size,
                              hipStream_t stream) {
    (void)in_sizes; (void)n_in; (void)d_ws; (void)ws_size; (void)out_size;

    const float* u   = (const float*)d_in[0];
    const float* y   = (const float*)d_in[1];
    const float* h0  = (const float*)d_in[2];
    const float* pw0 = (const float*)d_in[3];
    const float* pb0 = (const float*)d_in[4];
    const float* pw1 = (const float*)d_in[5];
    const float* pb1 = (const float*)d_in[6];
    const float* dw0 = (const float*)d_in[7];
    const float* db0 = (const float*)d_in[8];
    const float* dw1 = (const float*)d_in[9];
    const float* db1 = (const float*)d_in[10];
    const float* xmw = (const float*)d_in[11];
    const float* xmb = (const float*)d_in[12];
    const float* xlw = (const float*)d_in[13];
    const float* xlb = (const float*)d_in[14];
    const float* qw0 = (const float*)d_in[15];
    const float* qb0 = (const float*)d_in[16];
    const float* qw1 = (const float*)d_in[17];
    const float* qb1 = (const float*)d_in[18];
    const float* mw0 = (const float*)d_in[19];
    const float* mb0 = (const float*)d_in[20];
    const float* mw1 = (const float*)d_in[21];
    const float* mb1 = (const float*)d_in[22];
    const float* wih = (const float*)d_in[23];
    const float* whh = (const float*)d_in[24];
    const float* Cm  = (const float*)d_in[25];
    float* out = (float*)d_out;

    // opt-in to >64KB dynamic LDS (idempotent, deterministic)
    (void)hipFuncSetAttribute(reinterpret_cast<const void*>(vrnn_kernel),
                              hipFuncAttributeMaxDynamicSharedMemorySize, SMEM_BYTES);

    zero_out_kernel<<<1, 32, 0, stream>>>(out);
    vrnn_kernel<<<B_ / BM, NTHR, SMEM_BYTES, stream>>>(
        u, y, h0, pw0, pb0, pw1, pb1, dw0, db0, dw1, db1,
        xmw, xmb, xlw, xlb, qw0, qb0, qw1, qb1, mw0, mb0, mw1, mb1,
        wih, whh, Cm, out);
}
